// LoRALayer_86011015070181
// MI455X (gfx1250) — matrix operand order repaired and checked by reading, not compiled, not run on hardware
//
#include <hip/hip_runtime.h>
#include <hip/hip_bf16.h>
#include <stddef.h>
#include <stdint.h>

// ---- CDNA5 (gfx1250) vector types for WMMA and memory ops -------------------
typedef __attribute__((ext_vector_type(16))) __bf16 v16bf;  // WMMA A/B operand
typedef __attribute__((ext_vector_type(8)))  float  v8f;    // WMMA C/D operand
typedef __attribute__((ext_vector_type(4))) unsigned u32x4;
typedef __attribute__((ext_vector_type(2))) unsigned u32x2;
typedef __attribute__((ext_vector_type(4))) float    f32x4;
typedef __attribute__((ext_vector_type(4))) int      i32x4;

static constexpr int M_TOT = 16384;   // B*S = 4*4096
static constexpr int N_TOT = 4096;    // D_OUT
static constexpr int K_TOT = 4096;    // D_IN
static constexpr int RANK  = 8;
static constexpr float SCALING = 16.0f / 8.0f;

static constexpr int BM = 128;  // block M tile
static constexpr int BN = 128;  // block N tile
static constexpr int BK = 32;   // K step (== WMMA bf16 K)

// Hardware packed f32->bf16 conversion: one VALU op for two elements.
__device__ __forceinline__ unsigned cvt_pk_bf16(float lo, float hi) {
  unsigned r;
  asm("v_cvt_pk_bf16_f32 %0, %1, %2" : "=v"(r) : "v"(lo), "v"(hi));
  return r;
}

// Async global->LDS 16-byte copy (GLOBAL_LOAD_ASYNC_TO_LDS_B128, ASYNCcnt),
// with graceful fallback to a synchronous copy if the builtin is absent.
// Builtin signature (from probe): (v4i32 addrspace(1)*, v4i32 addrspace(3)*,
// imm offset, imm cpol).
__device__ __forceinline__ void async_copy_b128(const void* g, void* l) {
#if defined(__HIP_DEVICE_COMPILE__) && \
    __has_builtin(__builtin_amdgcn_global_load_async_to_lds_b128)
  __builtin_amdgcn_global_load_async_to_lds_b128(
      (__attribute__((address_space(1))) i32x4*)g,
      (__attribute__((address_space(3))) i32x4*)l, 0, 0);
#else
  *(u32x4*)l = *(const u32x4*)g;
#endif
}
__device__ __forceinline__ void wait_async0() {
#if defined(__HIP_DEVICE_COMPILE__) && \
    __has_builtin(__builtin_amdgcn_s_wait_asynccnt)
  __builtin_amdgcn_s_wait_asynccnt(0);
#elif defined(__HIP_DEVICE_COMPILE__)
  asm volatile("s_wait_asynccnt 0" ::: "memory");
#endif
}

// ---------------------------------------------------------------------------
// Kernel 1: L[m, r] = sum_k x[m,k] * A[r,k]      (tiny rank-8 down-projection)
// One wave32 per row m; lanes stripe K in float4 chunks; xor-shuffle reduce.
// ---------------------------------------------------------------------------
__global__ __launch_bounds__(256) void lora_low_kernel(
    const float* __restrict__ x, const float* __restrict__ A,
    float* __restrict__ L) {
  const int gtid = blockIdx.x * blockDim.x + threadIdx.x;
  const int row  = gtid >> 5;          // one wave per row
  const int lane = threadIdx.x & 31;

  const f32x4* x4 = (const f32x4*)(x + (size_t)row * K_TOT);
  const f32x4* A4 = (const f32x4*)A;

  float acc[RANK];
#pragma unroll
  for (int r = 0; r < RANK; ++r) acc[r] = 0.0f;

  for (int k4 = lane; k4 < K_TOT / 4; k4 += 32) {
    f32x4 xv = x4[k4];
#pragma unroll
    for (int r = 0; r < RANK; ++r) {
      f32x4 av = A4[(size_t)r * (K_TOT / 4) + k4];
      acc[r] += xv.x * av.x + xv.y * av.y + xv.z * av.z + xv.w * av.w;
    }
  }
#pragma unroll
  for (int off = 16; off >= 1; off >>= 1) {
#pragma unroll
    for (int r = 0; r < RANK; ++r)
      acc[r] += __shfl_xor(acc[r], off, 32);
  }
  if (lane < RANK) L[(size_t)row * RANK + lane] = acc[lane];
}

// ---------------------------------------------------------------------------
// Kernel 1b: W (f32, [N,K]) -> bf16 copy in workspace (read 128x by the GEMM).
// Each thread converts 8 elements: 2x b128 load, 4x v_cvt_pk, 1x b128 store.
// ---------------------------------------------------------------------------
__global__ __launch_bounds__(256) void convert_w_kernel(
    const float* __restrict__ w, unsigned* __restrict__ wbf) {
  const size_t tid = (size_t)blockIdx.x * 256 + threadIdx.x;
  const f32x4* w4 = (const f32x4*)w;
  f32x4 a = w4[tid * 2 + 0];
  f32x4 b = w4[tid * 2 + 1];
  u32x4 o = { cvt_pk_bf16(a.x, a.y), cvt_pk_bf16(a.z, a.w),
              cvt_pk_bf16(b.x, b.y), cvt_pk_bf16(b.z, b.w) };
  ((u32x4*)wbf)[tid] = o;
}

// ---------------------------------------------------------------------------
// Kernel 2: out = x @ W^T + SCALING * (L @ B^T), bf16 WMMA with f32 accum.
// Block: 256 thr (8 waves), tile 128x128; wave tile 32x64 (2x4 WMMA 16x16).
// Double-buffered LDS, ONE barrier per K-step. x converts on the fly
// (v_cvt_pk_bf16_f32); W streams async into LDS from a bf16 copy when
// ASYNC_W, else converts on the fly like x.
// ---------------------------------------------------------------------------
template <bool ASYNC_W>
__global__ __launch_bounds__(256) void lora_gemm_kernel(
    const float* __restrict__ x, const float* __restrict__ w,
    const unsigned short* __restrict__ wbf,
    const float* __restrict__ L, const float* __restrict__ lB,
    float* __restrict__ out) {
  __shared__ __align__(16) unsigned sx[2][BM][BK / 2];  // packed bf16 pairs
  __shared__ __align__(16) unsigned sw[2][BN][BK / 2];
  __shared__ __align__(16) float sL[BM][RANK];
  __shared__ __align__(16) float sB[BN][RANK];

  const int t    = threadIdx.x;
  const int lane = t & 31;
  const int wv   = t >> 5;
  const int wmb  = (wv & 3) * 32;   // wave M offset in block (4 waves in M)
  const int wnb  = (wv >> 2) * 64;  // wave N offset in block (2 waves in N)
  const int frow = lane & 15;       // fragment row (M for A, N for B/C/D)
  const int fsel = lane >> 4;       // K-half / M-half select

  const int gm0 = blockIdx.y * BM;
  const int gn0 = blockIdx.x * BN;

  // Stage LoRA epilogue operands via async global->LDS copies.
  {
    int row = t >> 1, c = (t & 1) * 4;  // 256 threads cover 128x8 floats
    async_copy_b128(&L[(size_t)(gm0 + row) * RANK + c], &sL[row][c]);
    async_copy_b128(&lB[(size_t)(gn0 + row) * RANK + c], &sB[row][c]);
  }

  v8f acc[2][4];
#pragma unroll
  for (int i = 0; i < 2; ++i)
#pragma unroll
    for (int j = 0; j < 4; ++j) acc[i][j] = (v8f){};

  // Per k-step each thread stages 4 float4 of x (and of w in sync mode):
  // 128 rows x 8 float4/row = 1024 float4 / 256 threads = 4 each.
  f32x4 rx[4], rw2[4];

  auto ldglobal = [&](int ks) {
    const int k0 = ks * BK;
#pragma unroll
    for (int j = 0; j < 4; ++j) {
      int f = t + 256 * j, row = f >> 3, c4 = f & 7;
      rx[j] = *(const f32x4*)&x[(size_t)(gm0 + row) * K_TOT + k0 + c4 * 4];
      if constexpr (!ASYNC_W)
        rw2[j] = *(const f32x4*)&w[(size_t)(gn0 + row) * K_TOT + k0 + c4 * 4];
    }
  };
  // Async: stream the bf16 W tile (128x32 bf16 = 8KB) straight into LDS.
  auto async_w_tile = [&](int ks, int bufT) {
#pragma unroll
    for (int j = 0; j < 2; ++j) {
      int f = t + 256 * j;           // 512 x 16B chunks
      int row = f >> 2, c = f & 3;   // 4 chunks per 64B row
      async_copy_b128(&wbf[(size_t)(gn0 + row) * K_TOT + ks * BK + c * 8],
                      &sw[bufT][row][c * 4]);
    }
  };
  auto stlds = [&](int buf) {
#pragma unroll
    for (int j = 0; j < 4; ++j) {
      int f = t + 256 * j, row = f >> 3, c4 = f & 7;
      u32x2 px = { cvt_pk_bf16(rx[j].x, rx[j].y), cvt_pk_bf16(rx[j].z, rx[j].w) };
      *(u32x2*)&sx[buf][row][c4 * 2] = px;   // ds_store_b64
      if constexpr (!ASYNC_W) {
        u32x2 pw = { cvt_pk_bf16(rw2[j].x, rw2[j].y), cvt_pk_bf16(rw2[j].z, rw2[j].w) };
        *(u32x2*)&sw[buf][row][c4 * 2] = pw;
      }
    }
  };

  const int KT = K_TOT / BK;
  ldglobal(0);
  if constexpr (ASYNC_W) async_w_tile(0, 0);
  stlds(0);
  wait_async0();       // sL/sB staging (+W tile 0 when ASYNC_W) complete
  __syncthreads();

  for (int ks = 0; ks < KT; ++ks) {
    const int buf = ks & 1;
    if (ks + 1 < KT) {
      ldglobal(ks + 1);                       // overlap global latency w/ WMMA
      if constexpr (ASYNC_W) async_w_tile(ks + 1, buf ^ 1);
      if (ks + 2 < KT) {                      // cache hints (global_prefetch_b8)
        __builtin_prefetch(&x[(size_t)(gm0 + (t >> 1)) * K_TOT + (ks + 2) * BK], 0, 1);
        if constexpr (!ASYNC_W)
          __builtin_prefetch(&w[(size_t)(gn0 + (t >> 1)) * K_TOT + (ks + 2) * BK], 0, 1);
      }
    }

    // A fragment (16x32 bf16): lane holds K[c0..c0+7] and K[c0+16..c0+23],
    // c0 = 8*fsel.  B fragment (32x16): same K layout as A.
    union Frag { u32x4 q[2]; v16bf v; } a[2], b[4];
#pragma unroll
    for (int i = 0; i < 2; ++i) {
      const unsigned* p = &sx[buf][wmb + i * 16 + frow][fsel * 4];
      a[i].q[0] = *(const u32x4*)p;
      a[i].q[1] = *(const u32x4*)(p + 8);
    }
#pragma unroll
    for (int j = 0; j < 4; ++j) {
      const unsigned* p = &sw[buf][wnb + j * 16 + frow][fsel * 4];
      b[j].q[0] = *(const u32x4*)p;
      b[j].q[1] = *(const u32x4*)(p + 8);
    }

#pragma unroll
    for (int i = 0; i < 2; ++i)
#pragma unroll
      for (int j = 0; j < 4; ++j)
        acc[i][j] = __builtin_amdgcn_wmma_f32_16x16x32_bf16(
            false, a[i].v, false, b[j].v, (short)0, acc[i][j], false, false);

    if (ks + 1 < KT) {
      // buf^1 readers all finished before the previous barrier -> safe to
      // fill it now; one barrier per K-step publishes it for the next step.
      stlds(buf ^ 1);
      if constexpr (ASYNC_W) wait_async0();
      __syncthreads();
    }
  }

  // Epilogue: D layout — VGPR i: M = i + 8*fsel, N = frow (per 16x16 tile).
  float bv[4][RANK];
#pragma unroll
  for (int j = 0; j < 4; ++j)
#pragma unroll
    for (int r = 0; r < RANK; ++r)
      bv[j][r] = sB[wnb + j * 16 + frow][r];

#pragma unroll
  for (int ti = 0; ti < 2; ++ti) {
#pragma unroll
    for (int i = 0; i < 8; ++i) {
      const int ml = wmb + ti * 16 + 8 * fsel + i;
      float lr[RANK];
#pragma unroll
      for (int r = 0; r < RANK; ++r) lr[r] = sL[ml][r];
      const size_t off = (size_t)(gm0 + ml) * N_TOT + gn0;
#pragma unroll
      for (int j = 0; j < 4; ++j) {
        float lora = 0.0f;
#pragma unroll
        for (int r = 0; r < RANK; ++r) lora += lr[r] * bv[j][r];
        out[off + wnb + j * 16 + frow] = acc[ti][j][i] + SCALING * lora;
      }
    }
  }
}

// ---------------------------------------------------------------------------
extern "C" void kernel_launch(void* const* d_in, const int* in_sizes, int n_in,
                              void* d_out, int out_size, void* d_ws, size_t ws_size,
                              hipStream_t stream) {
  (void)in_sizes; (void)n_in; (void)out_size;
  const float* x  = (const float*)d_in[0];  // [16384, 4096]
  const float* w  = (const float*)d_in[1];  // [4096, 4096]
  const float* lA = (const float*)d_in[2];  // [8, 4096]
  const float* lB = (const float*)d_in[3];  // [4096, 8]
  float* out = (float*)d_out;               // [16384, 4096]

  float* L = (float*)d_ws;                  // [16384, 8] = 512 KB
  const size_t l_bytes   = (size_t)M_TOT * RANK * sizeof(float);
  const size_t wbf_bytes = (size_t)N_TOT * K_TOT * 2;   // 32 MB bf16 W copy

  // Rank-8 down-projection: one wave per row.
  lora_low_kernel<<<M_TOT * 32 / 256, 256, 0, stream>>>(x, lA, L);

  dim3 grid(N_TOT / BN, M_TOT / BM);  // (32, 128)
  if (ws_size >= l_bytes + wbf_bytes) {
    // Pre-convert W to bf16 once; GEMM streams it async into LDS.
    unsigned short* wbf = (unsigned short*)((char*)d_ws + l_bytes);
    convert_w_kernel<<<(N_TOT * K_TOT / 8) / 256, 256, 0, stream>>>(
        w, (unsigned*)wbf);
    lora_gemm_kernel<true><<<grid, 256, 0, stream>>>(x, w, wbf, L, lB, out);
  } else {
    lora_gemm_kernel<false><<<grid, 256, 0, stream>>>(x, w, nullptr, L, lB, out);
  }
}
